// BinReLUConvBN_85555748536336
// MI455X (gfx1250) — matrix operand order repaired and checked by reading, not compiled, not run on hardware
//
#include <hip/hip_runtime.h>

typedef _Float16 v16h __attribute__((ext_vector_type(16)));
typedef _Float16 v8h  __attribute__((ext_vector_type(8)));
typedef float    v8f  __attribute__((ext_vector_type(8)));

#define NIMG 16
#define CIN  32
#define COUT 32
#define HH   256
#define WW   256
#define HW   (HH*WW)          // 65536
#define CHW  (CIN*HW)         // 2097152
#define KPRIME 288            // 9*32

// float workspace offsets (in floats), region starts at ws + 64MB
#define P_SUM   0             // [512] per-plane sum
#define P_SQ    512           // [512] per-plane sumsq
#define P_DAB   1024          // [512] per-plane sum|xh|
#define P_ASUM  1536          // [512] per-plane sum(|xh|*mask)
#define P_ACNT  2048          // [512] per-plane sum(mask)
#define O_SCALE 2560          // [32]
#define O_SHIFT 2592          // [32]
#define O_DTHR  2624          // [16]
#define O_ALPHA 2640          // [16]
#define F_REGION_FLOATS 4096  // pad to 16KB

// ---------------- Pass 1: per-channel sum / sumsq (deterministic partials) ---
__global__ void k_stats(const float* __restrict__ x, float* __restrict__ F) {
  __shared__ float s1[256], s2[256];
  int pb = blockIdx.x;                       // plane (n,c): 0..511
  const float4* p = (const float4*)(x + (size_t)pb * HW);
  float sum = 0.f, sq = 0.f;
  for (int i = threadIdx.x; i < HW/4; i += 256) {
    float4 v = p[i];
    sum += v.x + v.y + v.z + v.w;
    sq  += v.x*v.x + v.y*v.y + v.z*v.z + v.w*v.w;
  }
  s1[threadIdx.x] = sum; s2[threadIdx.x] = sq; __syncthreads();
  for (int o = 128; o > 0; o >>= 1) {
    if (threadIdx.x < o) { s1[threadIdx.x] += s1[threadIdx.x+o]; s2[threadIdx.x] += s2[threadIdx.x+o]; }
    __syncthreads();
  }
  if (threadIdx.x == 0) { F[P_SUM + pb] = s1[0]; F[P_SQ + pb] = s2[0]; }
}

__global__ void k_fin1(const float* __restrict__ gamma, const float* __restrict__ beta,
                       float* __restrict__ F) {
  int c = threadIdx.x;                       // 32 threads
  float cnt = (float)(NIMG * HW);
  float sum = 0.f, sq = 0.f;
  for (int n = 0; n < NIMG; n++) { sum += F[P_SUM + n*32 + c]; sq += F[P_SQ + n*32 + c]; }
  float mean = sum / cnt;
  float var  = sq / cnt - mean * mean;
  float inv  = rsqrtf(var + 1e-5f);
  float sc   = gamma[c] * inv;
  F[O_SCALE + c] = sc;
  F[O_SHIFT + c] = beta[c] - mean * sc;
}

// ---------------- Pass 2: per-image sum|xh| -> delta ------------------------
__global__ void k_delta(const float* __restrict__ x, float* __restrict__ F) {
  __shared__ float s1[256];
  int pb = blockIdx.x; int c = pb & 31;
  float sc = F[O_SCALE + c], sh = F[O_SHIFT + c];
  const float4* p = (const float4*)(x + (size_t)pb * HW);
  float sum = 0.f;
  for (int i = threadIdx.x; i < HW/4; i += 256) {
    float4 v = p[i];
    sum += fabsf(v.x*sc+sh) + fabsf(v.y*sc+sh) + fabsf(v.z*sc+sh) + fabsf(v.w*sc+sh);
  }
  s1[threadIdx.x] = sum; __syncthreads();
  for (int o = 128; o > 0; o >>= 1) {
    if (threadIdx.x < o) s1[threadIdx.x] += s1[threadIdx.x+o];
    __syncthreads();
  }
  if (threadIdx.x == 0) F[P_DAB + pb] = s1[0];
}

__global__ void k_fin2(const float* __restrict__ delta_I, float* __restrict__ F) {
  int n = threadIdx.x;                       // 16 threads
  float s = 0.f;
  for (int c = 0; c < CIN; c++) s += F[P_DAB + n*32 + c];
  F[O_DTHR + n] = delta_I[0] * s / (float)CHW;
}

// ---------------- Pass 3: per-image alpha -----------------------------------
__global__ void k_alpha(const float* __restrict__ x, float* __restrict__ F) {
  __shared__ float s1[256], s2[256];
  int pb = blockIdx.x; int c = pb & 31; int n = pb >> 5;
  float sc = F[O_SCALE + c], sh = F[O_SHIFT + c], d = F[O_DTHR + n];
  const float4* p = (const float4*)(x + (size_t)pb * HW);
  float sum = 0.f, cnt = 0.f;
  for (int i = threadIdx.x; i < HW/4; i += 256) {
    float4 v = p[i];
    float a;
    a = fabsf(v.x*sc+sh); if (a > d) { sum += a; cnt += 1.f; }
    a = fabsf(v.y*sc+sh); if (a > d) { sum += a; cnt += 1.f; }
    a = fabsf(v.z*sc+sh); if (a > d) { sum += a; cnt += 1.f; }
    a = fabsf(v.w*sc+sh); if (a > d) { sum += a; cnt += 1.f; }
  }
  s1[threadIdx.x] = sum; s2[threadIdx.x] = cnt; __syncthreads();
  for (int o = 128; o > 0; o >>= 1) {
    if (threadIdx.x < o) { s1[threadIdx.x] += s1[threadIdx.x+o]; s2[threadIdx.x] += s2[threadIdx.x+o]; }
    __syncthreads();
  }
  if (threadIdx.x == 0) { F[P_ASUM + pb] = s1[0]; F[P_ACNT + pb] = s2[0]; }
}

__global__ void k_fin3(float* __restrict__ F) {
  int n = threadIdx.x;                       // 16 threads
  float s = 0.f, c = 0.f;
  for (int i = 0; i < CIN; i++) { s += F[P_ASUM + n*32 + i]; c += F[P_ACNT + n*32 + i]; }
  F[O_ALPHA + n] = s / c;
}

// ---------------- Pass 4: ternary quantize, write t in NHWC f16 -------------
__global__ void k_quant(const float* __restrict__ x, const float* __restrict__ F,
                        _Float16* __restrict__ t) {
  int p = blockIdx.x * 256 + threadIdx.x;    // pixel (n,h,w): 0 .. 16*65536-1
  int n = p >> 16; int hw = p & 65535;
  float d = F[O_DTHR + n];
  union { _Float16 h[32]; uint4 u[4]; } q;
  const float* xp = x + (size_t)n * CHW + hw;
  #pragma unroll
  for (int c = 0; c < CIN; c++) {
    float xh = xp[(size_t)c * HW] * F[O_SCALE + c] + F[O_SHIFT + c];
    q.h[c] = (xh > d) ? (_Float16)1.0f : ((xh < -d) ? (_Float16)-1.0f : (_Float16)0.0f);
  }
  uint4* dst = (uint4*)(t + (size_t)p * 32);
  dst[0] = q.u[0]; dst[1] = q.u[1]; dst[2] = q.u[2]; dst[3] = q.u[3];
}

// ---------------- Weight prep: f32 OIHW -> f16 [co][(kh*3+kw)*32+ci] --------
__global__ void k_wprep(const float* __restrict__ cw, _Float16* __restrict__ wT) {
  for (int i = threadIdx.x; i < COUT * KPRIME; i += 256) {
    int co = i / KPRIME; int k = i % KPRIME; int s = k >> 5; int ci = k & 31;
    wT[i] = (_Float16)cw[(co * CIN + ci) * 9 + s];
  }
}

// ---------------- Implicit-GEMM 3x3 conv via WMMA + alpha + ReLU ------------
// Block: 256 thr (8 waves). Each block: one (n, h, 128-px half-row).
// Wave: 16 px x 32 co via two f32 accumulators; K = 288 = 9 WMMA k-steps.
__global__ void __launch_bounds__(256) k_conv(const _Float16* __restrict__ t,
                                              const _Float16* __restrict__ wT,
                                              const float* __restrict__ F,
                                              float* __restrict__ out) {
  __shared__ _Float16 wlds[COUT * KPRIME]   __attribute__((aligned(64)));   // 18 KB
  __shared__ _Float16 tlds[3 * 130 * 32]    __attribute__((aligned(64)));   // ~24.4 KB
  int b = blockIdx.x;
  int n = b >> 9;
  int h = (b >> 1) & 255;
  int wbase = (b & 1) << 7;
  int tid = threadIdx.x;

  { // stage full weight matrix
    const uint4* src = (const uint4*)wT;
    uint4* dst = (uint4*)wlds;
    for (int i = tid; i < (COUT * KPRIME * 2) / 16; i += 256) dst[i] = src[i];
  }
  // stage input halo tile [3 rows][130 cols][32 ci], zero-padded at borders
  for (int idx = tid; idx < 3 * 130; idx += 256) {
    int r = idx / 130, c = idx % 130;
    int hin = h - 1 + r;
    int win = wbase - 1 + c;
    uint4* dst = (uint4*)(tlds + (size_t)idx * 32);
    if (hin >= 0 && hin < HH && win >= 0 && win < WW) {
      const uint4* src = (const uint4*)(t + ((size_t)(n * HH + hin) * WW + win) * 32);
      dst[0] = src[0]; dst[1] = src[1]; dst[2] = src[2]; dst[3] = src[3];
    } else {
      uint4 z = make_uint4(0u, 0u, 0u, 0u);
      dst[0] = z; dst[1] = z; dst[2] = z; dst[3] = z;
    }
  }
  __syncthreads();

  int lane = tid & 31;
  int wv   = tid >> 5;
  int lp   = lane & 15;        // A: co%16 ; B/D: pixel (N) ; D: N
  int hp   = lane >> 4;        // half-wave selector
  int wl   = (wv << 4) + lp;   // local output column 0..127

  v8f c0 = {}; v8f c1 = {};
  #pragma unroll
  for (int kk = 0; kk < 9; kk++) {
    int kh = kk / 3, kw = kk % 3;
    // B fragment: lane holds col N=lp, K = 32*kk + 16*hp + i  -> ci contiguous
    v16h bfr = *(const v16h*)(tlds + ((size_t)(kh * 130 + wl + kw) * 32 + (hp << 4)));
    // A fragments: lane holds row co, K = 32*kk + 16*hp + {0..15}
    int kbase = (kk << 5) + (hp << 4);
    v8h a0l = *(const v8h*)(wlds + lp * KPRIME + kbase);
    v8h a0h = *(const v8h*)(wlds + lp * KPRIME + kbase + 8);
    v8h a1l = *(const v8h*)(wlds + (16 + lp) * KPRIME + kbase);
    v8h a1h = *(const v8h*)(wlds + (16 + lp) * KPRIME + kbase + 8);
    v16h a0 = __builtin_shufflevector(a0l, a0h, 0,1,2,3,4,5,6,7,8,9,10,11,12,13,14,15);
    v16h a1 = __builtin_shufflevector(a1l, a1h, 0,1,2,3,4,5,6,7,8,9,10,11,12,13,14,15);
    c0 = __builtin_amdgcn_wmma_f32_16x16x32_f16(false, a0, false, bfr, (short)0, c0, false, false);
    c1 = __builtin_amdgcn_wmma_f32_16x16x32_f16(false, a1, false, bfr, (short)0, c1, false, false);
  }

  float alpha = F[O_ALPHA + n];
  int wout = wbase + wl;
  #pragma unroll
  for (int r = 0; r < 8; r++) {
    int co = r + (hp << 3);
    float v0 = c0[r] * alpha; v0 = v0 > 0.f ? v0 : 0.f;
    float v1 = c1[r] * alpha; v1 = v1 > 0.f ? v1 : 0.f;
    out[(((size_t)n * COUT + co)      * HH + h) * WW + wout] = v0;
    out[(((size_t)n * COUT + co + 16) * HH + h) * WW + wout] = v1;
  }
}

extern "C" void kernel_launch(void* const* d_in, const int* in_sizes, int n_in,
                              void* d_out, int out_size, void* d_ws, size_t ws_size,
                              hipStream_t stream) {
  const float* x       = (const float*)d_in[0];
  const float* gamma   = (const float*)d_in[1];
  const float* beta    = (const float*)d_in[2];
  const float* delta_I = (const float*)d_in[3];
  const float* conv_w  = (const float*)d_in[4];
  float* out = (float*)d_out;

  // ws layout: [t: 64MB f16 NHWC][F: 16KB floats][wT: 18KB f16]
  char* ws = (char*)d_ws;
  _Float16* t  = (_Float16*)ws;
  const size_t T_BYTES = (size_t)NIMG * HW * CIN * sizeof(_Float16); // 67108864
  float* F = (float*)(ws + T_BYTES);
  _Float16* wT = (_Float16*)(ws + T_BYTES + F_REGION_FLOATS * sizeof(float));

  k_stats<<<512, 256, 0, stream>>>(x, F);
  k_fin1 <<<1, 32, 0, stream>>>(gamma, beta, F);
  k_delta<<<512, 256, 0, stream>>>(x, F);
  k_fin2 <<<1, 16, 0, stream>>>(delta_I, F);
  k_alpha<<<512, 256, 0, stream>>>(x, F);
  k_fin3 <<<1, 16, 0, stream>>>(F);
  k_wprep<<<1, 256, 0, stream>>>(conv_w, wT);
  k_quant<<<4096, 256, 0, stream>>>(x, F, t);
  k_conv <<<NIMG * HH * 2, 256, 0, stream>>>(t, wT, F, out);
}
